// GNN_67697274520247
// MI455X (gfx1250) — matrix operand order repaired and checked by reading, not compiled, not run on hardware
//
#include <hip/hip_runtime.h>
#include <hip/hip_bf16.h>

// ---------------------------------------------------------------------------
// GNN message-passing network for MI455X (gfx1250, wave32, WMMA).
// All GEMMs run on v_wmma_f32_16x16x32_f16 (f16 A/B, f32 accumulate).
// One workgroup (256 threads = 8 wave32 waves) processes 16 rows; wave w owns
// output columns [16w, 16w+16).  Weights are pre-packed into B-fragment-major
// f16 blocks so each lane loads its b-frag as one contiguous 32B global load.
// SiLU uses v_exp_f32 + v_rcp_f32; A-tile staging uses float4 (b128) loads.
// ---------------------------------------------------------------------------

typedef __attribute__((ext_vector_type(16))) _Float16 v16h;
typedef __attribute__((ext_vector_type(8)))  _Float16 v8h;
typedef __attribute__((ext_vector_type(4)))  _Float16 v4h;
typedef __attribute__((ext_vector_type(8)))  float    v8f;

#define B_  2
#define N_  10000
#define E_  160000
#define H_  128
#define BN_ (B_ * N_)   // 20000  (divisible by 16)
#define BE_ (B_ * E_)   // 320000 (divisible by 16)

// Fast SiLU: x * sigmoid(x) with hardware reciprocal (v_rcp_f32).
__device__ __forceinline__ float silu_fast(float x) {
  return x * __builtin_amdgcn_rcpf(1.0f + __expf(-x));
}

__device__ __forceinline__ v4h cvt4(float4 v) {
  v4h r = {(_Float16)v.x, (_Float16)v.y, (_Float16)v.z, (_Float16)v.w};
  return r;
}

// ----------------------------- weight packing ------------------------------
// Fragment-major layout: for weight W[K,N] (row-major, in-dim x out-dim),
// block(kc, nt) holds the 32x16 sub-matrix for WMMA chunk kc, col-tile nt,
// as 512 halves, lane-major: lane l gets 16 contiguous halves, half j being
// W[kc*32 + (j<8 ? 8*(l>>4)+j : 16+8*(l>>4)+j-8)][nt*16 + (l&15)]  (ISA B-matrix 32x16 layout).
__global__ __launch_bounds__(256) void pack_weight_kernel(
    const float* __restrict__ w, _Float16* __restrict__ dst,
    int Ksrc, int Kpad, int N) {
  int t = blockIdx.x * 256 + threadIdx.x;
  int total = Kpad * N;
  if (t >= total) return;
  int blk    = t >> 9;        // 512 halves per 32x16 block
  int within = t & 511;
  int lane   = within >> 4;
  int j      = within & 15;
  int ntiles = N >> 4;
  int kc = blk / ntiles;
  int nt = blk - kc * ntiles;
  int k = kc * 32 + ((j < 8) ? (((lane >> 4) << 3) + j) : (16 + ((lane >> 4) << 3) + (j - 8)));
  int n = nt * 16 + (lane & 15);
  float v = (k < Ksrc) ? w[k * N + n] : 0.0f;
  dst[t] = (_Float16)v;
}

// ------------------------------ WMMA helpers -------------------------------
// A-fragment (16x32 f16) from an LDS tile stored row-major [16][strideH].
// Lane l: row m = l&15, group g = l>>4; elements 0..7 -> K = 8g+j,
// elements 8..15 -> K = 16+8g+j  (ISA 16-bit A 16x32 layout).
__device__ __forceinline__ v16h load_a_frag(const _Float16* __restrict__ lA,
                                            int strideH, int m, int g, int kc) {
  const _Float16* p = lA + m * strideH + kc * 32 + (g << 3);
  v8h lo = *(const v8h*)(p);
  v8h hi = *(const v8h*)(p + 16);
  v16h a;
#pragma unroll
  for (int i = 0; i < 8; ++i) { a[i] = lo[i]; a[i + 8] = hi[i]; }
  return a;
}

// 16-row x 16-col WMMA GEMM over K = KIN, B from packed weights.
template <int KIN, int NTILES>
__device__ __forceinline__ v8f wmma_rows16(const _Float16* __restrict__ lA,
                                           const _Float16* __restrict__ wp,
                                           int nt, int lane) {
  const int g = lane >> 4;
  const int m = lane & 15;
  v8f c = {};
#pragma unroll
  for (int kc = 0; kc < KIN / 32; ++kc) {
    v16h a = load_a_frag(lA, KIN, m, g, kc);
    v16h b = *(const v16h*)(wp + (((kc * NTILES + nt) << 9) + (lane << 4)));
    c = __builtin_amdgcn_wmma_f32_16x16x32_f16(false, a, false, b,
                                               (short)0, c, false, false);
  }
  return c;
}

// Two-layer MLP [KIN -> 128 (SiLU) -> 128] on one 16-row tile.
// Returns this wave's 16x16 C fragment of layer-2 (bias NOT yet added).
template <int KIN>
__device__ __forceinline__ v8f mlp2_128(const _Float16* __restrict__ lA,
                                        _Float16* __restrict__ lAct,
                                        const _Float16* __restrict__ w1p,
                                        const float* __restrict__ b1,
                                        const _Float16* __restrict__ w2p,
                                        const float* __restrict__ b2,
                                        float& bias2_out) {
  const int tid  = threadIdx.x;
  const int wave = tid >> 5;          // wave32: 8 waves per block
  const int lane = tid & 31;
  const int g    = lane >> 4;
  const int col  = (wave << 4) + (lane & 15);
  // layer 1
  v8f c1 = wmma_rows16<KIN, 8>(lA, w1p, wave, lane);
  const float bb1 = b1[col];
#pragma unroll
  for (int v = 0; v < 8; ++v) {
    float sv = silu_fast(c1[v] + bb1);
    lAct[(v + (g << 3)) * 128 + col] = (_Float16)sv;
  }
  __syncthreads();
  // layer 2
  v8f c2 = wmma_rows16<128, 8>(lAct, w2p, wave, lane);
  bias2_out = b2[col];
  return c2;
}

// ------------------------------- kernels -----------------------------------

__global__ __launch_bounds__(256) void zero_kernel(float* __restrict__ p, int n) {
  int i = blockIdx.x * 256 + threadIdx.x;
  if (i < n) p[i] = 0.0f;
}

// Encoder (node or edge): in[BR,inw] -> MLP[32(pad) -> 128 -> 128] -> out.
__global__ __launch_bounds__(256) void enc_kernel(
    const float* __restrict__ in, int inw,
    const _Float16* __restrict__ w1p, const float* __restrict__ b1,
    const _Float16* __restrict__ w2p, const float* __restrict__ b2,
    float* __restrict__ out) {
  __shared__ __align__(32) _Float16 lA[16 * 32];
  __shared__ __align__(32) _Float16 lAct[16 * 128];
  const int tile = blockIdx.x;
  const int t    = threadIdx.x;
  if (t < 128) {                       // 16 rows x 8 float4-slots (32-wide pad)
    int row = t >> 3, q = t & 7;
    long long r = (long long)tile * 16 + row;
    v4h hv = {};
    if ((q << 2) < inw) hv = cvt4(*(const float4*)(in + r * inw + (q << 2)));
    *(v4h*)(&lA[row * 32 + (q << 2)]) = hv;
  }
  __syncthreads();
  float bias2;
  v8f c2 = mlp2_128<32>(lA, lAct, w1p, b1, w2p, b2, bias2);
  const int lane = t & 31, g = lane >> 4;
  const int col  = ((t >> 5) << 4) + (lane & 15);
#pragma unroll
  for (int v = 0; v < 8; ++v) {
    long long r = (long long)tile * 16 + v + (g << 3);
    out[r * 128 + col] = c2[v] + bias2;
  }
}

// Edge update: A = concat(e, h[src], h[dest]) [16x384] -> MLP -> e_new.
// Epilogue: e += e_new; atomic scatter-add e_new into agg[dest].
__global__ __launch_bounds__(256) void edge_update_kernel(
    const int* __restrict__ ei,            // [2,E] (src row 0, dest row 1)
    const float* __restrict__ h,
    float* __restrict__ e, float* __restrict__ agg,
    const _Float16* __restrict__ w1p, const float* __restrict__ b1,
    const _Float16* __restrict__ w2p, const float* __restrict__ b2) {
  __shared__ __align__(32) _Float16 lA[16 * 384];
  __shared__ __align__(32) _Float16 lAct[16 * 128];
  __shared__ int lDst[16];
  const int tile = blockIdx.x;
  const int t    = threadIdx.x;
  {
    const int row = t >> 4;                // 16 threads stage each 384-f row
    const int sub = t & 15;
    const long long ge = (long long)tile * 16 + row;
    const int b   = (int)(ge / E_);
    const int eid = (int)(ge - (long long)b * E_);
    const int sN = ei[eid];
    const int dN = ei[E_ + eid];
    if (sub == 0) lDst[row] = b * N_ + dN;
    const float4* s0 = (const float4*)(e + ge * 128);
    const float4* s1 = (const float4*)(h + (long long)(b * N_ + sN) * 128);
    const float4* s2 = (const float4*)(h + (long long)(b * N_ + dN) * 128);
#pragma unroll
    for (int i = 0; i < 6; ++i) {
      int q   = sub * 6 + i;               // float4 slot within row [0,96)
      int seg = q >> 5, off = q & 31;      // 32 float4 per 128-f segment
      const float4* p = (seg == 0) ? s0 : (seg == 1) ? s1 : s2;
      *(v4h*)(&lA[row * 384 + (q << 2)]) = cvt4(p[off]);
    }
  }
  __syncthreads();
  float bias2;
  v8f c2 = mlp2_128<384>(lA, lAct, w1p, b1, w2p, b2, bias2);
  const int lane = t & 31, g = lane >> 4;
  const int col  = ((t >> 5) << 4) + (lane & 15);
#pragma unroll
  for (int v = 0; v < 8; ++v) {
    const int rowl = v + (g << 3);
    const long long ge = (long long)tile * 16 + rowl;
    const float val = c2[v] + bias2;                 // e_new element
    e[ge * 128 + col] += val;                        // e = e + e_new
    atomicAdd(&agg[(long long)lDst[rowl] * 128 + col], val);  // scatter-add
  }
}

// Node update: A = concat(h, agg) [16x256] -> MLP -> h += h_new.
__global__ __launch_bounds__(256) void node_update_kernel(
    const float* __restrict__ agg, float* __restrict__ h,
    const _Float16* __restrict__ w1p, const float* __restrict__ b1,
    const _Float16* __restrict__ w2p, const float* __restrict__ b2) {
  __shared__ __align__(32) _Float16 lA[16 * 256];
  __shared__ __align__(32) _Float16 lAct[16 * 128];
  const int tile = blockIdx.x;
  const int t    = threadIdx.x;
  {
    const int row = t >> 4;
    const int sub = t & 15;
    const long long r = (long long)tile * 16 + row;
    const float4* s0 = (const float4*)(h + r * 128);
    const float4* s1 = (const float4*)(agg + r * 128);
#pragma unroll
    for (int i = 0; i < 4; ++i) {
      int q   = sub * 4 + i;               // float4 slot within row [0,64)
      int seg = q >> 5, off = q & 31;
      const float4* p = seg ? s1 : s0;
      *(v4h*)(&lA[row * 256 + (q << 2)]) = cvt4(p[off]);
    }
  }
  __syncthreads();
  float bias2;
  v8f c2 = mlp2_128<256>(lA, lAct, w1p, b1, w2p, b2, bias2);
  const int lane = t & 31, g = lane >> 4;
  const int col  = ((t >> 5) << 4) + (lane & 15);
#pragma unroll
  for (int v = 0; v < 8; ++v) {
    long long r = (long long)tile * 16 + v + (g << 3);
    h[r * 128 + col] += c2[v] + bias2;               // residual
  }
}

// Decoder: h [16x128] -> 128 (SiLU) -> 16.
__global__ __launch_bounds__(256) void dec_kernel(
    const float* __restrict__ h,
    const _Float16* __restrict__ w1p, const float* __restrict__ b1,
    const _Float16* __restrict__ w2p, const float* __restrict__ b2,
    float* __restrict__ out) {
  __shared__ __align__(32) _Float16 lA[16 * 128];
  __shared__ __align__(32) _Float16 lAct[16 * 128];
  const int tile = blockIdx.x;
  const int t    = threadIdx.x;
  {
    const int row = t >> 4;
    const int sub = t & 15;
    const long long r = (long long)tile * 16 + row;
    const float4* s0 = (const float4*)(h + r * 128);
#pragma unroll
    for (int i = 0; i < 2; ++i) {
      int q = sub * 2 + i;                 // float4 slot within row [0,32)
      *(v4h*)(&lA[row * 128 + (q << 2)]) = cvt4(s0[q]);
    }
  }
  __syncthreads();
  const int wave = t >> 5, lane = t & 31, g = lane >> 4;
  const int col  = (wave << 4) + (lane & 15);
  v8f c1 = wmma_rows16<128, 8>(lA, w1p, wave, lane);
  const float bb1 = b1[col];
#pragma unroll
  for (int v = 0; v < 8; ++v) {
    float sv = silu_fast(c1[v] + bb1);
    lAct[(v + (g << 3)) * 128 + col] = (_Float16)sv;
  }
  __syncthreads();
  if (wave == 0) {                       // layer 2 output is only 16 wide
    v8f c2 = wmma_rows16<128, 1>(lAct, w2p, 0, lane);
    const float bb2 = b2[lane & 15];
#pragma unroll
    for (int v = 0; v < 8; ++v) {
      long long r = (long long)tile * 16 + v + (g << 3);
      out[r * 16 + (lane & 15)] = c2[v] + bb2;
    }
  }
}

// ------------------------------ host launch --------------------------------
// d_in order: 0:x 1:edge_index 2:edge_attr
//   3..6:  enc_node  W0[16,128] b0 W1[128,128] b1
//   7..10: enc_edge  W0[8,128]  b0 W1[128,128] b1
//   11+8s: step s:   edgeW0[384,128] eb0 edgeW1[128,128] eb1
//                    nodeW0[256,128] nb0 nodeW1[128,128] nb1
//   35..38: dec W0[128,128] b0 W1[128,16] b1
//
// ws: h (10.24MB) | agg (10.24MB) | e (163.84MB) | packed f16 weights (~0.8MB)
extern "C" void kernel_launch(void* const* d_in, const int* in_sizes, int n_in,
                              void* d_out, int out_size, void* d_ws, size_t ws_size,
                              hipStream_t stream) {
  (void)in_sizes; (void)n_in; (void)out_size; (void)ws_size;
  const float* x  = (const float*)d_in[0];
  const int*   ei = (const int*)d_in[1];     // per harness: integer -> int*
  const float* ea = (const float*)d_in[2];

  char* ws = (char*)d_ws;
  float*    h   = (float*)(ws);
  float*    agg = (float*)(ws + 10240000);
  float*    e   = (float*)(ws + 20480000);
  _Float16* wp  = (_Float16*)(ws + 184320000);

  // packed-weight offsets (in halves)
  const size_t O_NW1 = 0,      O_NW2 = 4096;
  const size_t O_EW1 = 20480,  O_EW2 = 24576;
  const size_t O_STEP = 40960, STEP_SZ = 114688;
  const size_t S_EW1 = 0, S_EW2 = 49152, S_NW1 = 65536, S_NW2 = 98304;
  const size_t O_DW1 = 385024, O_DW2 = 401408;

  auto packW = [&](const float* w, size_t offH, int Ksrc, int Kpad, int N) {
    int total = Kpad * N;
    pack_weight_kernel<<<(total + 255) / 256, 256, 0, stream>>>(
        w, wp + offH, Ksrc, Kpad, N);
  };
  packW((const float*)d_in[3], O_NW1, 16, 32, 128);
  packW((const float*)d_in[5], O_NW2, 128, 128, 128);
  packW((const float*)d_in[7], O_EW1, 8, 32, 128);
  packW((const float*)d_in[9], O_EW2, 128, 128, 128);
  for (int s = 0; s < 3; ++s) {
    size_t base = O_STEP + (size_t)s * STEP_SZ;
    packW((const float*)d_in[11 + 8 * s + 0], base + S_EW1, 384, 384, 128);
    packW((const float*)d_in[11 + 8 * s + 2], base + S_EW2, 128, 128, 128);
    packW((const float*)d_in[11 + 8 * s + 4], base + S_NW1, 256, 256, 128);
    packW((const float*)d_in[11 + 8 * s + 6], base + S_NW2, 128, 128, 128);
  }
  packW((const float*)d_in[35], O_DW1, 128, 128, 128);
  packW((const float*)d_in[37], O_DW2, 128, 128, 16);

  // encoders
  enc_kernel<<<BN_ / 16, 256, 0, stream>>>(
      x, 16, wp + O_NW1, (const float*)d_in[4], wp + O_NW2,
      (const float*)d_in[6], h);
  enc_kernel<<<BE_ / 16, 256, 0, stream>>>(
      ea, 8, wp + O_EW1, (const float*)d_in[8], wp + O_EW2,
      (const float*)d_in[10], e);

  // message-passing steps
  for (int s = 0; s < 3; ++s) {
    size_t base = O_STEP + (size_t)s * STEP_SZ;
    zero_kernel<<<(BN_ * 128 + 255) / 256, 256, 0, stream>>>(agg, BN_ * 128);
    edge_update_kernel<<<BE_ / 16, 256, 0, stream>>>(
        ei, h, e, agg,
        wp + base + S_EW1, (const float*)d_in[11 + 8 * s + 1],
        wp + base + S_EW2, (const float*)d_in[11 + 8 * s + 3]);
    node_update_kernel<<<BN_ / 16, 256, 0, stream>>>(
        agg, h,
        wp + base + S_NW1, (const float*)d_in[11 + 8 * s + 5],
        wp + base + S_NW2, (const float*)d_in[11 + 8 * s + 7]);
  }

  // decoder
  dec_kernel<<<BN_ / 16, 256, 0, stream>>>(
      h, wp + O_DW1, (const float*)d_in[36], wp + O_DW2,
      (const float*)d_in[38], (float*)d_out);
}
